// GemNetOCBackbone_67568425501310
// MI455X (gfx1250) — matrix operand order repaired and checked by reading, not compiled, not run on hardware
//
#include <hip/hip_runtime.h>
#include <stdint.h>

// ---------------------------------------------------------------------------
// GemNet-OC backbone for MI455X (gfx1250, wave32, WMMA bf16 16x16x32).
// Dense layers: one WMMA GEMM kernel, 64x64 block tile, B panel staged into
// LDS via the Tensor Data Mover (tensor_load_to_lds + s_wait_tensorcnt),
// 4 WMMA tiles per wave with A-fragment reuse, fused SiLU/residual epilogue.
// Triplet bilinear: gather-WMMA kernel with LDS f32 atomics for the cbf
// contraction and global f32 atomics for the segment sum.
// ---------------------------------------------------------------------------

typedef __bf16 bf16_t;
typedef __attribute__((ext_vector_type(16))) __bf16 v16bf;
typedef __attribute__((ext_vector_type(8)))  __bf16 v8bf;
typedef __attribute__((ext_vector_type(8)))  float  v8f;
typedef __attribute__((ext_vector_type(4)))  unsigned int u32x4;
typedef __attribute__((ext_vector_type(8)))  int i32x8;
typedef __attribute__((ext_vector_type(4)))  int i32x4;

#define INV_SQRT2F 0.70710678118654752440f

static __device__ __forceinline__ float silu_f(float v) {
  return v / (1.0f + __expf(-v));
}

// ---------------------------------------------------------------------------
// GEMM: out = epilogue( A[rows,K](bf16) @ Bt[ncols,K]^T (bf16) )
// Block: 128 threads = 4 waves; block tile 64 rows x 64 cols.
// Wave w: rows r0+w*16..+15, all 64 cols -> 4 accumulators; A fragment
// loaded once per 32-K step (global b128), reused across 4 B fragments
// read from the TDM-staged LDS B panel (ds_load_b128).
// grid = (ceil(rows/64), ncols/64); dynamic LDS = 64*K*2 bytes.
// ---------------------------------------------------------------------------
__launch_bounds__(128)
__global__ void k_gemm_bf16(const bf16_t* __restrict__ A, int lda,
                            const bf16_t* __restrict__ Bt, int K, int rows,
                            int do_silu,
                            const float* __restrict__ resid, int ldr,
                            float scale,
                            float* __restrict__ outF, int ldoF,
                            bf16_t* __restrict__ outB, int ldoB) {
  extern __shared__ char smem[];
  bf16_t* Bs = (bf16_t*)smem;  // [64][K] staged B panel

  const int tid  = threadIdx.x;
  const int wid  = tid >> 5;
  const int lane = tid & 31;
  const int r0   = blockIdx.x * 64 + wid * 16;
  const int n0   = blockIdx.y * 64;

  // ---- TDM: stage B panel (64 contiguous Bt rows = 64*K bf16) into LDS ----
  if (tid < 32) {
    const unsigned long long gaddr =
        (unsigned long long)(uintptr_t)(Bt + (size_t)n0 * K);
    const unsigned int ldsa = (unsigned int)(uintptr_t)Bs;
    u32x4 g0;
    g0[0] = 1u;                                   // count=1 (valid descriptor)
    g0[1] = ldsa;                                 // lds_addr
    g0[2] = (unsigned int)(gaddr & 0xFFFFFFFFull);
    g0[3] = (unsigned int)((gaddr >> 32) & 0x1FFFFFFull) | (2u << 30);  // type=2
    i32x8 g1;
    g1[0] = 0x00010000;                 // data_size=1 (2 bytes/elem)
    g1[1] = (int)((K & 0xFFFF) << 16);  // tensor_dim0 lo16
    g1[2] = (int)(64 << 16);            // tensor_dim0 hi | tensor_dim1=64 lo
    g1[3] = (int)((K & 0xFFFF) << 16);  // tensor_dim1 hi | tile_dim0=K
    g1[4] = 64;                         // tile_dim1=64 (tile_dim2=0)
    g1[5] = K;                          // tensor_dim0_stride lo32
    g1[6] = 0;
    g1[7] = 0;
    i32x4 gz4 = {0, 0, 0, 0};
    i32x8 gz8 = {0, 0, 0, 0, 0, 0, 0, 0};
    __builtin_amdgcn_tensor_load_to_lds(g0, g1, gz4, gz4, gz8, 0);
    __builtin_amdgcn_s_wait_tensorcnt(0);
  }
  __syncthreads();

  const int ahalf = (lane < 16) ? 0 : 8;    // A fragment K-offset per lane
  const int khalf = (lane < 16) ? 0 : 8;    // B fragment K-offset per lane
  const int frow  = lane & 15;
  const int bcol  = lane & 15;

  int arow = r0 + frow;
  if (arow >= rows) arow = rows - 1;        // clamp for the ragged last block
  const size_t abase = (size_t)arow * lda;

  v8f acc[4];
#pragma unroll
  for (int ct = 0; ct < 4; ++ct)
    acc[ct] = (v8f){0.f, 0.f, 0.f, 0.f, 0.f, 0.f, 0.f, 0.f};

  for (int k0 = 0; k0 < K; k0 += 32) {
    // A fragment (ISA 7.12.2: 16-bit A 16x32)
    v8bf alo = *reinterpret_cast<const v8bf*>(&A[abase + k0 + ahalf]);
    v8bf ahi = *reinterpret_cast<const v8bf*>(&A[abase + k0 + ahalf + 16]);
    v16bf afrag;
#pragma unroll
    for (int i = 0; i < 8; ++i) { afrag[i] = alo[i]; afrag[8 + i] = ahi[i]; }
    if (k0 + 32 < K)  // global_prefetch_b8 of next A chunk
      __builtin_prefetch(&A[abase + k0 + 32], 0, 1);

#pragma unroll
    for (int ct = 0; ct < 4; ++ct) {
      const size_t boff = (size_t)(ct * 16 + bcol) * K + k0 + khalf;
      v8bf b0 = *reinterpret_cast<const v8bf*>(&Bs[boff]);
      v8bf b1 = *reinterpret_cast<const v8bf*>(&Bs[boff + 16]);
      v16bf bfrag;
#pragma unroll
      for (int i = 0; i < 8; ++i) { bfrag[i] = b0[i]; bfrag[8 + i] = b1[i]; }
      acc[ct] = __builtin_amdgcn_wmma_f32_16x16x32_bf16(
          false, afrag, false, bfrag, (short)0, acc[ct], false, false);
    }
  }

  // Epilogue (C/D layout: VGPR g -> row g + (lane<16?0:8), col = lane&15)
  const int orow0 = r0 + ((lane < 16) ? 0 : 8);
#pragma unroll
  for (int ct = 0; ct < 4; ++ct) {
    const int ocol = n0 + ct * 16 + (lane & 15);
#pragma unroll
    for (int g = 0; g < 8; ++g) {
      const int orow = orow0 + g;
      if (orow < rows) {
        float v = acc[ct][g];
        if (do_silu) v = silu_f(v);
        if (resid)   v = resid[(size_t)orow * ldr + ocol] + v;
        v *= scale;
        if (outF) outF[(size_t)orow * ldoF + ocol] = v;
        if (outB) outB[(size_t)orow * ldoB + ocol] = (bf16_t)v;
      }
    }
  }
}

// ---------------------------------------------------------------------------
// Triplet bilinear: per 16 triplets, y[t,o] = sum_c cb[t,c] *
//   (x[id3_ba[t]] @ Wbil[c,:,o]).  Wbt packed as [c*64+o][i] bf16.
// ---------------------------------------------------------------------------
__launch_bounds__(128)
__global__ void k_triplet(const bf16_t* __restrict__ xbf,   // [E,64]
                          const float*  __restrict__ cb,    // [T,16]
                          const bf16_t* __restrict__ Wbt,   // [1024,64]
                          const int* __restrict__ id3_ba,
                          const int* __restrict__ id3_ca,
                          float* __restrict__ agg) {        // [E,64]
  __shared__ bf16_t As[16 * 64];
  __shared__ float  cbs[16 * 16];
  __shared__ float  ys[16 * 64];

  const int tid  = threadIdx.x;
  const int wid  = tid >> 5;
  const int lane = tid & 31;
  const int t0   = blockIdx.x * 16;

  {  // gather 16 rows of x[id3_ba] into LDS
    const int r = tid >> 3, cchunk = (tid & 7) * 8;
    const int e = id3_ba[t0 + r];
    *reinterpret_cast<v8bf*>(&As[r * 64 + cchunk]) =
        *reinterpret_cast<const v8bf*>(&xbf[(size_t)e * 64 + cchunk]);
  }
  cbs[tid]       = cb[(size_t)t0 * 16 + tid];
  cbs[tid + 128] = cb[(size_t)t0 * 16 + tid + 128];
  for (int i = tid; i < 1024; i += 128) ys[i] = 0.f;
  __syncthreads();

  const int khalf = (lane < 16) ? 0 : 8;
  const int ahalf = (lane < 16) ? 0 : 8;
  const int frow  = lane & 15;
  const int trow0 = (lane < 16) ? 0 : 8;

  for (int gi = 0; gi < 16; ++gi) {
    const int g = wid * 16 + gi;
    const int n = g * 16 + (lane & 15);   // flattened (c,o): c=n>>6, o=n&63
    v8f acc = {0.f, 0.f, 0.f, 0.f, 0.f, 0.f, 0.f, 0.f};
#pragma unroll
    for (int kk = 0; kk < 64; kk += 32) {
      v8bf alo = *reinterpret_cast<const v8bf*>(&As[frow * 64 + kk + ahalf]);
      v8bf ahi = *reinterpret_cast<const v8bf*>(&As[frow * 64 + kk + ahalf + 16]);
      v16bf afrag;
#pragma unroll
      for (int i = 0; i < 8; ++i) { afrag[i] = alo[i]; afrag[8 + i] = ahi[i]; }
      const size_t boff = (size_t)n * 64 + kk + khalf;
      v8bf b0 = *reinterpret_cast<const v8bf*>(&Wbt[boff]);
      v8bf b1 = *reinterpret_cast<const v8bf*>(&Wbt[boff + 16]);
      v16bf bfrag;
#pragma unroll
      for (int i = 0; i < 8; ++i) { bfrag[i] = b0[i]; bfrag[8 + i] = b1[i]; }
      acc = __builtin_amdgcn_wmma_f32_16x16x32_bf16(
          false, afrag, false, bfrag, (short)0, acc, false, false);
    }
    const int c = n >> 6, o = n & 63;
#pragma unroll
    for (int q = 0; q < 8; ++q) {
      const int tr = trow0 + q;
      atomicAdd(&ys[tr * 64 + o], acc[q] * cbs[tr * 16 + c]);  // ds_add_f32
    }
  }
  __syncthreads();

  for (int i = tid; i < 1024; i += 128) {
    const int tr = i >> 6, o = i & 63;
    const int e = id3_ca[t0 + tr];
    atomicAdd(&agg[(size_t)e * 64 + o], ys[i]);  // global_atomic_add_f32
  }
}

// ------------------------------ small kernels ------------------------------

__global__ void k_wt(const float* __restrict__ W, bf16_t* __restrict__ Bt,
                     int K, int Ncols) {
  const long i = (long)blockIdx.x * 256 + threadIdx.x;
  if (i >= (long)K * Ncols) return;
  const int k = (int)(i / Ncols), n = (int)(i % Ncols);
  Bt[(size_t)n * K + k] = (bf16_t)W[i];
}

__global__ void k_bilpack(const float* __restrict__ Wb, bf16_t* __restrict__ Bt) {
  const int i = blockIdx.x * 256 + threadIdx.x;  // 16*64*64 = 65536
  if (i >= 65536) return;
  const int c = i >> 12, ii = (i >> 6) & 63, o = i & 63;
  Bt[(size_t)(c * 64 + o) * 64 + ii] = (bf16_t)Wb[i];
}

__global__ void k_gather_h(const int* __restrict__ z,
                           const float* __restrict__ table,
                           float* __restrict__ h, bf16_t* __restrict__ hcat,
                           long total) {
  const long i = (long)blockIdx.x * 256 + threadIdx.x;
  if (i >= total) return;
  const int n = (int)(i >> 7), a = (int)(i & 127);
  const float v = table[(size_t)z[n] * 128 + a];
  h[i] = v;
  hcat[(size_t)n * 384 + a] = (bf16_t)v;
}

__global__ void k_feat(const int* __restrict__ idx_s, const int* __restrict__ idx_t,
                       const float* __restrict__ h, const float* __restrict__ rbf,
                       bf16_t* __restrict__ feat, long total) {
  const long i = (long)blockIdx.x * 256 + threadIdx.x;
  if (i >= total) return;
  const int e = (int)(i / 384), j = (int)(i % 384);
  float v;
  if (j < 128)      v = h[(size_t)idx_s[e] * 128 + j];
  else if (j < 256) v = h[(size_t)idx_t[e] * 128 + (j - 128)];
  else              v = rbf[(size_t)e * 128 + (j - 256)];
  feat[i] = (bf16_t)v;
}

__global__ void k_cbf(const float* __restrict__ cbf, const float* __restrict__ Wc,
                      float* __restrict__ cb, long total) {
  const long i = (long)blockIdx.x * 256 + threadIdx.x;
  if (i >= total) return;
  const int t = (int)(i >> 4), c = (int)(i & 15);
  float s = 0.f;
#pragma unroll
  for (int j = 0; j < 7; ++j) s += cbf[(size_t)t * 7 + j] * Wc[j * 16 + c];
  cb[i] = s;
}

__global__ void k_zero_f32(float* __restrict__ p, long n) {
  const long i = (long)blockIdx.x * 256 + threadIdx.x;
  if (i < n) p[i] = 0.f;
}

__global__ void k_cvt(const float* __restrict__ a, bf16_t* __restrict__ o, long n) {
  const long i = (long)blockIdx.x * 256 + threadIdx.x;
  if (i < n) o[i] = (bf16_t)a[i];
}

__global__ void k_mulcvt(const float* __restrict__ a, const float* __restrict__ b,
                         bf16_t* __restrict__ o, long n) {
  const long i = (long)blockIdx.x * 256 + threadIdx.x;
  if (i < n) o[i] = (bf16_t)(a[i] * b[i]);
}

__global__ void k_combine(float* __restrict__ x, const float* __restrict__ y,
                          float s, bf16_t* __restrict__ xb, long n) {
  const long i = (long)blockIdx.x * 256 + threadIdx.x;
  if (i >= n) return;
  const float v = (x[i] + y[i]) * s;
  x[i] = v;
  xb[i] = (bf16_t)v;
}

__global__ void k_addh(float* __restrict__ h, const float* __restrict__ r,
                       bf16_t* __restrict__ hcat, int off, long n) {
  const long i = (long)blockIdx.x * 256 + threadIdx.x;
  if (i >= n) return;
  const float v = h[i] + r[i];
  h[i] = v;
  const int a = (int)(i & 127);
  hcat[(size_t)(i >> 7) * 384 + off + a] = (bf16_t)v;
}

__global__ void k_copy_slice(const bf16_t* __restrict__ src, bf16_t* __restrict__ dst,
                             int off, long total) {  // src ld 256 -> dst ld 768
  const long i = (long)blockIdx.x * 256 + threadIdx.x;
  if (i >= total) return;
  dst[(size_t)(i >> 8) * 768 + off + (int)(i & 255)] = src[i];
}

__global__ void k_scatter_xa(const float* __restrict__ xa, const int* __restrict__ idx_t,
                             float* __restrict__ ha, long total) {
  const long i = (long)blockIdx.x * 256 + threadIdx.x;
  if (i >= total) return;
  const int e = (int)(i >> 7), a = (int)(i & 127);
  atomicAdd(&ha[(size_t)idx_t[e] * 128 + a], xa[i]);
}

__global__ void k_energy(const float* __restrict__ xe, const float* __restrict__ wfin,
                         float* __restrict__ out, int N) {
  const int n = blockIdx.x * 256 + threadIdx.x;
  if (n >= N) return;
  float s = 0.f;
  for (int k = 0; k < 128; ++k) s += xe[(size_t)n * 128 + k] * wfin[k];
  out[n * 4 + 0] = s;
  out[n * 4 + 1] = 0.f; out[n * 4 + 2] = 0.f; out[n * 4 + 3] = 0.f;
}

__global__ void k_forces(const float* __restrict__ xf, const float* __restrict__ wfin,
                         const int* __restrict__ idx_t, const float* __restrict__ V,
                         float* __restrict__ out, int E) {
  const int e = blockIdx.x * 256 + threadIdx.x;
  if (e >= E) return;
  float s = 0.f;
  for (int k = 0; k < 256; ++k) s += xf[(size_t)e * 256 + k] * wfin[k];
  const int n = idx_t[e];
#pragma unroll
  for (int d = 0; d < 3; ++d)
    atomicAdd(&out[n * 4 + 1 + d], s * V[(size_t)e * 3 + d]);
}

// ------------------------------ host driver --------------------------------

extern "C" void kernel_launch(void* const* d_in, const int* in_sizes, int n_in,
                              void* d_out, int out_size, void* d_ws, size_t ws_size,
                              hipStream_t stream) {
  (void)n_in; (void)out_size; (void)ws_size;
  const int N = in_sizes[0];
  const int E = in_sizes[1];
  const int T = in_sizes[3];

  const int*   atomic_numbers = (const int*)d_in[0];
  const int*   idx_s     = (const int*)d_in[1];
  const int*   idx_t     = (const int*)d_in[2];
  const int*   id3_ba    = (const int*)d_in[3];
  const int*   id3_ca    = (const int*)d_in[4];
  const float* rbf       = (const float*)d_in[5];
  const float* cbf       = (const float*)d_in[6];
  const float* V_st      = (const float*)d_in[7];
  const float* atom_table= (const float*)d_in[8];
  const float* W_edge_in = (const float*)d_in[9];
  const float* rbf_w3    = (const float*)d_in[10];
  const float* W_down    = (const float*)d_in[11];
  const float* W_cbf     = (const float*)d_in[12];
  const float* W_bil     = (const float*)d_in[13];
  const float* W_up      = (const float*)d_in[14];
  const float* W_res_e   = (const float*)d_in[15];
  const float* rbf_wh    = (const float*)d_in[16];
  const float* W_e2a     = (const float*)d_in[17];
  const float* W_res_a   = (const float*)d_in[18];
  const float* W_outE_in = (const float*)d_in[19];
  const float* W_outE_res= (const float*)d_in[20];
  const float* W_outE_fin= (const float*)d_in[21];
  const float* W_outF_in = (const float*)d_in[22];
  const float* W_outF_res= (const float*)d_in[23];
  const float* W_outF_fin= (const float*)d_in[24];
  float* out = (float*)d_out;

  char* wp = (char*)d_ws;
  auto bump = [&](size_t bytes) -> void* {
    void* r = (void*)wp;
    wp += (bytes + 255) & ~(size_t)255;
    return r;
  };
  auto CD = [](long n) -> unsigned { return (unsigned)((n + 255) / 256); };

  // ---- activation buffers ----
  float*  h     = (float*) bump((size_t)N * 128 * 4);
  bf16_t* hcat  = (bf16_t*)bump((size_t)N * 384 * 2);
  bf16_t* rbfB  = (bf16_t*)bump((size_t)E * 128 * 2);
  bf16_t* feat  = (bf16_t*)bump((size_t)E * 384 * 2);
  float*  mF    = (float*) bump((size_t)E * 256 * 4);
  bf16_t* mBF   = (bf16_t*)bump((size_t)E * 256 * 2);
  bf16_t* Fcat  = (bf16_t*)bump((size_t)E * 768 * 2);
  float*  P1    = (float*) bump((size_t)E * 256 * 4);
  bf16_t* SBF   = (bf16_t*)bump((size_t)E * 256 * 2);
  bf16_t* XBF   = (bf16_t*)bump((size_t)E * 64 * 2);
  float*  CB    = (float*) bump((size_t)T * 16 * 4);
  float*  AGG   = (float*) bump((size_t)E * 64 * 4);
  bf16_t* AGGBF = (bf16_t*)bump((size_t)E * 64 * 2);
  bf16_t* T1BF  = (bf16_t*)bump((size_t)E * 256 * 2);
  float*  M2F   = (float*) bump((size_t)E * 256 * 4);
  bf16_t* M2BF  = (bf16_t*)bump((size_t)E * 256 * 2);
  float*  XA    = (float*) bump((size_t)E * 128 * 4);
  float*  HA    = (float*) bump((size_t)N * 128 * 4);
  bf16_t* HABF  = (bf16_t*)bump((size_t)N * 128 * 2);
  bf16_t* TN1BF = (bf16_t*)bump((size_t)N * 128 * 2);
  float*  RN    = (float*) bump((size_t)N * 128 * 4);
  float*  XE1F  = (float*) bump((size_t)N * 128 * 4);
  bf16_t* XE1BF = (bf16_t*)bump((size_t)N * 128 * 2);
  float*  XEF   = (float*) bump((size_t)N * 128 * 4);

  // ---- bf16 transposed weights ----
  bf16_t* WedgeT = (bf16_t*)bump(384 * 256 * 2);
  bf16_t *Rbf3T[2], *WdownT[2], *WbilT[2], *WupT[2], *WresET[2][4],
         *RbfhT[2], *We2aT[2], *WresAT[2][2];
  for (int b = 0; b < 2; ++b) {
    Rbf3T[b] = (bf16_t*)bump(128 * 256 * 2);
    WdownT[b] = (bf16_t*)bump(256 * 64 * 2);
    WbilT[b] = (bf16_t*)bump(1024 * 64 * 2);
    WupT[b] = (bf16_t*)bump(64 * 256 * 2);
    for (int j = 0; j < 4; ++j) WresET[b][j] = (bf16_t*)bump(256 * 256 * 2);
    RbfhT[b] = (bf16_t*)bump(128 * 256 * 2);
    We2aT[b] = (bf16_t*)bump(256 * 128 * 2);
    for (int j = 0; j < 2; ++j) WresAT[b][j] = (bf16_t*)bump(128 * 128 * 2);
  }
  bf16_t* WoutEinT   = (bf16_t*)bump(384 * 128 * 2);
  bf16_t* WoutEresT0 = (bf16_t*)bump(128 * 128 * 2);
  bf16_t* WoutEresT1 = (bf16_t*)bump(128 * 128 * 2);
  bf16_t* WoutFinT   = (bf16_t*)bump(768 * 256 * 2);
  bf16_t* WoutFresT0 = (bf16_t*)bump(256 * 256 * 2);
  bf16_t* WoutFresT1 = (bf16_t*)bump(256 * 256 * 2);

  auto gemm = [&](const bf16_t* A, int lda, const bf16_t* Bt, int K,
                  int rows, int ncols, int silu, const float* resid, int ldr,
                  float scale, float* outF, int ldoF, bf16_t* outB, int ldoB) {
    dim3 grid((unsigned)((rows + 63) / 64), (unsigned)(ncols / 64));
    size_t shm = (size_t)64 * K * 2;  // B panel in LDS
    k_gemm_bf16<<<grid, 128, shm, stream>>>(A, lda, Bt, K, rows, silu, resid,
                                            ldr, scale, outF, ldoF, outB, ldoB);
  };

  // ---- weight preparation (transpose + bf16) ----
  k_wt<<<CD(384 * 256), 256, 0, stream>>>(W_edge_in, WedgeT, 384, 256);
  for (int b = 0; b < 2; ++b) {
    k_wt<<<CD(128 * 256), 256, 0, stream>>>(rbf_w3 + (size_t)b * 128 * 256, Rbf3T[b], 128, 256);
    k_wt<<<CD(256 * 64), 256, 0, stream>>>(W_down + (size_t)b * 256 * 64, WdownT[b], 256, 64);
    k_bilpack<<<CD(65536), 256, 0, stream>>>(W_bil + (size_t)b * 65536, WbilT[b]);
    k_wt<<<CD(64 * 256), 256, 0, stream>>>(W_up + (size_t)b * 64 * 256, WupT[b], 64, 256);
    for (int j = 0; j < 4; ++j)
      k_wt<<<CD(256 * 256), 256, 0, stream>>>(W_res_e + (size_t)(b * 4 + j) * 256 * 256,
                                              WresET[b][j], 256, 256);
    k_wt<<<CD(128 * 256), 256, 0, stream>>>(rbf_wh + (size_t)b * 128 * 256, RbfhT[b], 128, 256);
    k_wt<<<CD(256 * 128), 256, 0, stream>>>(W_e2a + (size_t)b * 256 * 128, We2aT[b], 256, 128);
    for (int j = 0; j < 2; ++j)
      k_wt<<<CD(128 * 128), 256, 0, stream>>>(W_res_a + (size_t)(b * 2 + j) * 128 * 128,
                                              WresAT[b][j], 128, 128);
  }
  k_wt<<<CD(384 * 128), 256, 0, stream>>>(W_outE_in, WoutEinT, 384, 128);
  k_wt<<<CD(128 * 128), 256, 0, stream>>>(W_outE_res, WoutEresT0, 128, 128);
  k_wt<<<CD(128 * 128), 256, 0, stream>>>(W_outE_res + 128 * 128, WoutEresT1, 128, 128);
  k_wt<<<CD(768 * 256), 256, 0, stream>>>(W_outF_in, WoutFinT, 768, 256);
  k_wt<<<CD(256 * 256), 256, 0, stream>>>(W_outF_res, WoutFresT0, 256, 256);
  k_wt<<<CD(256 * 256), 256, 0, stream>>>(W_outF_res + 256 * 256, WoutFresT1, 256, 256);

  // ---- embeddings ----
  k_gather_h<<<CD((long)N * 128), 256, 0, stream>>>(atomic_numbers, atom_table, h, hcat,
                                                    (long)N * 128);
  k_cvt<<<CD((long)E * 128), 256, 0, stream>>>(rbf, rbfB, (long)E * 128);
  k_feat<<<CD((long)E * 384), 256, 0, stream>>>(idx_s, idx_t, h, rbf, feat, (long)E * 384);
  // m = silu(feat @ W_edge_in)
  gemm(feat, 384, WedgeT, 384, E, 256, 1, nullptr, 0, 1.0f, mF, 256, mBF, 256);
  k_copy_slice<<<CD((long)E * 256), 256, 0, stream>>>(mBF, Fcat, 0, (long)E * 256);

  for (int b = 0; b < 2; ++b) {
    // x = (m * (rbf@rbf_w3)) @ W_down
    gemm(rbfB, 128, Rbf3T[b], 128, E, 256, 0, nullptr, 0, 1.0f, P1, 256, nullptr, 0);
    k_mulcvt<<<CD((long)E * 256), 256, 0, stream>>>(mF, P1, SBF, (long)E * 256);
    gemm(SBF, 256, WdownT[b], 256, E, 64, 0, nullptr, 0, 1.0f, nullptr, 0, XBF, 64);
    // cb = cbf @ W_cbf
    k_cbf<<<CD((long)T * 16), 256, 0, stream>>>(cbf, W_cbf + (size_t)b * 7 * 16, CB, (long)T * 16);
    // agg = segsum over triplets of bilinear
    k_zero_f32<<<CD((long)E * 64), 256, 0, stream>>>(AGG, (long)E * 64);
    k_triplet<<<(unsigned)(T / 16), 128, 0, stream>>>(XBF, CB, WbilT[b], id3_ba, id3_ca, AGG);
    // x3 = silu(agg @ W_up)
    k_cvt<<<CD((long)E * 64), 256, 0, stream>>>(AGG, AGGBF, (long)E * 64);
    gemm(AGGBF, 64, WupT[b], 64, E, 256, 1, nullptr, 0, 1.0f, P1, 256, nullptr, 0);
    // m2 = residual(m)
    gemm(mBF, 256, WresET[b][0], 256, E, 256, 1, nullptr, 0, 1.0f, nullptr, 0, T1BF, 256);
    gemm(T1BF, 256, WresET[b][1], 256, E, 256, 1, mF, 256, INV_SQRT2F, M2F, 256, nullptr, 0);
    // m2 = (m2 + x3)/sqrt2
    k_combine<<<CD((long)E * 256), 256, 0, stream>>>(M2F, P1, INV_SQRT2F, M2BF, (long)E * 256);
    // m = residual(m2)
    gemm(M2BF, 256, WresET[b][2], 256, E, 256, 1, nullptr, 0, 1.0f, nullptr, 0, T1BF, 256);
    gemm(T1BF, 256, WresET[b][3], 256, E, 256, 1, M2F, 256, INV_SQRT2F, mF, 256, mBF, 256);
    // atom update: ha = segsum((m * (rbf@rbf_wh)) @ W_e2a)
    gemm(rbfB, 128, RbfhT[b], 128, E, 256, 0, nullptr, 0, 1.0f, P1, 256, nullptr, 0);
    k_mulcvt<<<CD((long)E * 256), 256, 0, stream>>>(mF, P1, SBF, (long)E * 256);
    gemm(SBF, 256, We2aT[b], 256, E, 128, 0, nullptr, 0, 1.0f, XA, 128, nullptr, 0);
    k_zero_f32<<<CD((long)N * 128), 256, 0, stream>>>(HA, (long)N * 128);
    k_scatter_xa<<<CD((long)E * 128), 256, 0, stream>>>(XA, idx_t, HA, (long)E * 128);
    // h += residual(ha)
    k_cvt<<<CD((long)N * 128), 256, 0, stream>>>(HA, HABF, (long)N * 128);
    gemm(HABF, 128, WresAT[b][0], 128, N, 128, 1, nullptr, 0, 1.0f, nullptr, 0, TN1BF, 128);
    gemm(TN1BF, 128, WresAT[b][1], 128, N, 128, 1, HA, 128, INV_SQRT2F, RN, 128, nullptr, 0);
    k_addh<<<CD((long)N * 128), 256, 0, stream>>>(h, RN, hcat, (b + 1) * 128, (long)N * 128);
    k_copy_slice<<<CD((long)E * 256), 256, 0, stream>>>(mBF, Fcat, (b + 1) * 256, (long)E * 256);
  }

  // ---- energy head ----
  gemm(hcat, 384, WoutEinT, 384, N, 128, 1, nullptr, 0, 1.0f, XE1F, 128, XE1BF, 128);
  gemm(XE1BF, 128, WoutEresT0, 128, N, 128, 1, nullptr, 0, 1.0f, nullptr, 0, TN1BF, 128);
  gemm(TN1BF, 128, WoutEresT1, 128, N, 128, 1, XE1F, 128, INV_SQRT2F, XEF, 128, nullptr, 0);
  k_energy<<<CD(N), 256, 0, stream>>>(XEF, W_outE_fin, out, N);

  // ---- force head ----
  gemm(Fcat, 768, WoutFinT, 768, E, 256, 1, nullptr, 0, 1.0f, P1, 256, SBF, 256);
  gemm(SBF, 256, WoutFresT0, 256, E, 256, 1, nullptr, 0, 1.0f, nullptr, 0, T1BF, 256);
  gemm(T1BF, 256, WoutFresT1, 256, E, 256, 1, P1, 256, INV_SQRT2F, M2F, 256, nullptr, 0);
  k_forces<<<CD(E), 256, 0, stream>>>(M2F, W_outF_fin, idx_t, V_st, out, E);
}
